// GAT_79955111182791
// MI455X (gfx1250) — matrix operand order repaired and checked by reading, not compiled, not run on hardware
//
#include <hip/hip_runtime.h>
#include <hip/hip_bf16.h>

#define NN   20000
#define EE   160000
#define FIN  128
#define DD   128
#define HH   4
#define HD   512     // HH * DD
#define NEGS 0.2f

typedef __attribute__((ext_vector_type(16))) __bf16 v16bf;
typedef __attribute__((ext_vector_type(8)))  __bf16 v8bf;
typedef __attribute__((ext_vector_type(8)))  float  v8f;

__device__ __forceinline__ unsigned fenc(float f) {
  unsigned u = __float_as_uint(f);
  return (u & 0x80000000u) ? ~u : (u | 0x80000000u);
}
__device__ __forceinline__ float fdec(unsigned e) {
  return (e & 0x80000000u) ? __uint_as_float(e ^ 0x80000000u) : __uint_as_float(~e);
}

__global__ void k_zero(float* p, size_t n) {
  size_t i = (size_t)blockIdx.x * blockDim.x + threadIdx.x;
  if (i < n) p[i] = 0.0f;
}

// accumulate incoming edge_attr sums and counts per destination node
__global__ void k_ea_scatter(const int* __restrict__ dstI, const float* __restrict__ ea,
                             float* __restrict__ easum, float* __restrict__ cnt) {
  int e = blockIdx.x * blockDim.x + threadIdx.x;
  if (e >= EE) return;
  int d = dstI[e];
  atomicAdd(&easum[d * 2 + 0], ea[e * 2 + 0]);
  atomicAdd(&easum[d * 2 + 1], ea[e * 2 + 1]);
  atomicAdd(&cnt[d], 1.0f);
}

__global__ void k_loop_div(float* __restrict__ easum, const float* __restrict__ cnt) {
  int n = blockIdx.x * blockDim.x + threadIdx.x;
  if (n >= NN) return;
  float c = fmaxf(cnt[n], 1.0f);
  easum[n * 2 + 0] /= c;
  easum[n * 2 + 1] /= c;
}

__global__ void k_f2bf(const float* __restrict__ s, __bf16* __restrict__ d, int n) {
  int i = blockIdx.x * blockDim.x + threadIdx.x;
  if (i < n) d[i] = (__bf16)s[i];
}

// W row-major fp32 [K][Nout] -> Wt bf16 column-major [Nout][K]
__global__ void k_wt_bf(const float* __restrict__ W, __bf16* __restrict__ Wt, int K, int Nout) {
  int i = blockIdx.x * blockDim.x + threadIdx.x;
  if (i >= K * Nout) return;
  int n = i / K, k = i - n * K;
  Wt[i] = (__bf16)W[(size_t)k * Nout + n];
}

// C[M][Nout] = A[M][K] (bf16 row-major) x Bt[Nout][K] (bf16 col-major) + bias[Nout]
// One wave per 16x16 tile; v_wmma_f32_16x16x32_bf16, K/32 steps.
__global__ void k_wmma_gemm(const __bf16* __restrict__ A, const __bf16* __restrict__ Bt,
                            const float* __restrict__ bias, float* __restrict__ C,
                            int M, int Nout, int K) {
  int wave = (int)(((size_t)blockIdx.x * blockDim.x + threadIdx.x) >> 5);
  int lane = threadIdx.x & 31;
  int ntN = Nout >> 4;
  int tiles = (M >> 4) * ntN;
  if (wave >= tiles) return;
  int mt = wave / ntN, nt = wave - mt * ntN;
  int half = lane >> 4, lq = lane & 15;

  const __bf16* pa = A + (size_t)(mt * 16 + lq) * K;            // A: row M=lq
  const __bf16* pb = Bt + (size_t)(nt * 16 + lq) * K + half * 16; // B: col N=lq, K half

  float bv = bias[nt * 16 + lq];
  v8f c;
#pragma unroll
  for (int j = 0; j < 8; ++j) c[j] = bv;

  int nk = K >> 5;
  for (int kt = 0; kt < nk; ++kt) {
    int k0 = kt << 5;
    // A layout: lanes 0-15 K=0..15, lanes 16-31 K=16..31 (matches B)
    v8bf a0 = *(const v8bf*)(pa + k0 + half * 16);
    v8bf a1 = *(const v8bf*)(pa + k0 + half * 16 + 8);
    v16bf a = __builtin_shufflevector(a0, a1, 0, 1, 2, 3, 4, 5, 6, 7,
                                      8, 9, 10, 11, 12, 13, 14, 15);
    // B layout: lanes 0-15 K=0..15, lanes 16-31 K=16..31 (contiguous in col-major)
    v16bf b = *(const v16bf*)(pb + k0);
    c = __builtin_amdgcn_wmma_f32_16x16x32_bf16(false, a, false, b, (short)0, c, false, false);
  }
  // C/D layout: VGPR j, lane -> (M = j + 8*half, N = lq)
  float* pc = C + (size_t)(mt * 16 + half * 8) * Nout + nt * 16 + lq;
#pragma unroll
  for (int j = 0; j < 8; ++j) pc[(size_t)j * Nout] = c[j];
}

// Pass 1: per-edge logits (recompute ee = fea@We inline, K=2) + atomic segment max.
// One wave per edge; lane owns 16 features; 8 lanes per head.
__global__ void k_edge_logits(const float* __restrict__ xl, const float* __restrict__ xr,
                              const float* __restrict__ ea, const float* __restrict__ lattr,
                              const int* __restrict__ srcI, const int* __restrict__ dstI,
                              const float* __restrict__ We, const float* __restrict__ att,
                              float* __restrict__ logits, unsigned* __restrict__ mEnc) {
  int wave = (int)(((size_t)blockIdx.x * blockDim.x + threadIdx.x) >> 5);
  int lane = threadIdx.x & 31;
  const int nE = EE + NN;
  if (wave >= nE) return;
  int s, d;
  float f0, f1;
  if (wave < EE) {
    s = srcI[wave]; d = dstI[wave];
    f0 = ea[(size_t)wave * 2]; f1 = ea[(size_t)wave * 2 + 1];
  } else {
    s = d = wave - EE;
    f0 = lattr[(size_t)(wave - EE) * 2]; f1 = lattr[(size_t)(wave - EE) * 2 + 1];
  }
  const float* pl = xl + (size_t)s * HD + lane * 16;
  const float* pr = xr + (size_t)d * HD + lane * 16;
  const float* w0 = We + lane * 16;
  const float* w1 = We + HD + lane * 16;
  const float* pt = att + lane * 16;
  float acc = 0.f;
#pragma unroll
  for (int i = 0; i < 16; i += 4) {
    float4 a = *(const float4*)(pl + i);
    float4 b = *(const float4*)(pr + i);
    float4 u = *(const float4*)(w0 + i);
    float4 v = *(const float4*)(w1 + i);
    float4 t = *(const float4*)(pt + i);
    float e;
    e = a.x + b.x + f0 * u.x + f1 * v.x; e = e > 0.f ? e : NEGS * e; acc += e * t.x;
    e = a.y + b.y + f0 * u.y + f1 * v.y; e = e > 0.f ? e : NEGS * e; acc += e * t.y;
    e = a.z + b.z + f0 * u.z + f1 * v.z; e = e > 0.f ? e : NEGS * e; acc += e * t.z;
    e = a.w + b.w + f0 * u.w + f1 * v.w; e = e > 0.f ? e : NEGS * e; acc += e * t.w;
  }
  acc += __shfl_xor(acc, 1);
  acc += __shfl_xor(acc, 2);
  acc += __shfl_xor(acc, 4);
  if ((lane & 7) == 0) {
    int h = lane >> 3;
    logits[(size_t)wave * HH + h] = acc;
    atomicMax(&mEnc[d * HH + h], fenc(acc));
  }
}

// Pass 2: w = exp(logit - segmax), accumulate denominator
__global__ void k_edge_expw(const int* __restrict__ dstI, float* __restrict__ lw,
                            const unsigned* __restrict__ mEnc, float* __restrict__ denom) {
  int i = blockIdx.x * blockDim.x + threadIdx.x;
  const int tot = (EE + NN) * HH;
  if (i >= tot) return;
  int e = i >> 2, h = i & 3;
  int d = (e < EE) ? dstI[e] : (e - EE);
  float m = fdec(mEnc[d * HH + h]);
  float w = __expf(lw[i] - m);
  lw[i] = w;
  atomicAdd(&denom[d * HH + h], w);
}

// Pass 3: msg = xl[src] * alpha, scatter-add into out[dst]
__global__ void k_edge_scatter(const float* __restrict__ xl, const float* __restrict__ lw,
                               const float* __restrict__ denom,
                               const int* __restrict__ srcI, const int* __restrict__ dstI,
                               float* __restrict__ outacc) {
  int wave = (int)(((size_t)blockIdx.x * blockDim.x + threadIdx.x) >> 5);
  int lane = threadIdx.x & 31;
  const int nE = EE + NN;
  if (wave >= nE) return;
  int s, d;
  if (wave < EE) { s = srcI[wave]; d = dstI[wave]; }
  else           { s = d = wave - EE; }
  int h = lane >> 3;
  float alpha = lw[(size_t)wave * HH + h] / denom[d * HH + h];
  const float* pl = xl + (size_t)s * HD + lane * 16;
  float* po = outacc + (size_t)d * HD + lane * 16;
#pragma unroll
  for (int i = 0; i < 16; ++i) atomicAdd(&po[i], pl[i] * alpha);
}

// Node update: mean over heads + bias_out, leaky_relu(0.01)
__global__ void k_node(const float* __restrict__ outacc, const float* __restrict__ bias,
                       float* __restrict__ hout) {
  int i = blockIdx.x * blockDim.x + threadIdx.x;
  if (i >= NN * DD) return;
  int n = i >> 7, dd = i & 127;
  const float* p = outacc + (size_t)n * HD;
  float v = (p[dd] + p[DD + dd] + p[2 * DD + dd] + p[3 * DD + dd]) * 0.25f + bias[dd];
  hout[i] = v > 0.f ? v : 0.01f * v;
}

extern "C" void kernel_launch(void* const* d_in, const int* in_sizes, int n_in,
                              void* d_out, int out_size, void* d_ws, size_t ws_size,
                              hipStream_t stream) {
  (void)in_sizes; (void)n_in; (void)out_size; (void)ws_size;
  const float* x    = (const float*)d_in[0];
  const int*   ei   = (const int*)d_in[1];
  const float* ea   = (const float*)d_in[2];
  const float* Wf   = (const float*)d_in[3];
  const float* bf_  = (const float*)d_in[4];
  const float* Wl   = (const float*)d_in[5];
  const float* bl   = (const float*)d_in[6];
  const float* Wr   = (const float*)d_in[7];
  const float* br   = (const float*)d_in[8];
  const float* We   = (const float*)d_in[9];
  const float* att  = (const float*)d_in[10];
  const float* bout = (const float*)d_in[11];
  const int* srcI = ei;
  const int* dstI = ei + EE;

  // carve workspace (256B aligned)
  char* p = (char*)d_ws;
  auto carve = [&](size_t bytes) -> void* {
    void* r = (void*)p;
    p += (bytes + 255) & ~(size_t)255;
    return r;
  };
  float*    h32   = (float*)carve((size_t)NN * DD * 4);
  __bf16*   hbf   = (__bf16*)carve((size_t)NN * DD * 2);
  __bf16*   WfT   = (__bf16*)carve((size_t)DD * FIN * 2);
  __bf16*   WlT   = (__bf16*)carve((size_t)HD * DD * 2);
  __bf16*   WrT   = (__bf16*)carve((size_t)HD * DD * 2);
  float*    xl    = (float*)carve((size_t)NN * HD * 4);
  float*    xr    = (float*)carve((size_t)NN * HD * 4);
  float*    lattr = (float*)carve((size_t)NN * 2 * 4);
  float*    cnt   = (float*)carve((size_t)NN * 4);
  float*    lw    = (float*)carve((size_t)(EE + NN) * HH * 4);
  unsigned* mEnc  = (unsigned*)carve((size_t)NN * HH * 4);
  float*    denom = (float*)carve((size_t)NN * HH * 4);
  float*    outac = (float*)carve((size_t)NN * HD * 4);

  const int B = 256;
  auto gz = [&](size_t n) { return (unsigned)((n + B - 1) / B); };

  // self-loop edge_attr = mean of incoming edge_attr
  k_zero<<<gz((size_t)NN * 2), B, 0, stream>>>(lattr, (size_t)NN * 2);
  k_zero<<<gz((size_t)NN), B, 0, stream>>>(cnt, (size_t)NN);
  k_ea_scatter<<<gz((size_t)EE), B, 0, stream>>>(dstI, ea, lattr, cnt);
  k_loop_div<<<gz((size_t)NN), B, 0, stream>>>(lattr, cnt);

  // weights -> bf16 transposed (col-major) for the WMMA B operand
  k_wt_bf<<<gz((size_t)FIN * DD), B, 0, stream>>>(Wf, WfT, FIN, DD);
  k_wt_bf<<<gz((size_t)DD * HD), B, 0, stream>>>(Wl, WlT, DD, HD);
  k_wt_bf<<<gz((size_t)DD * HD), B, 0, stream>>>(Wr, WrT, DD, HD);

  // h = x @ Wf + bf
  k_f2bf<<<gz((size_t)NN * FIN), B, 0, stream>>>(x, hbf, NN * FIN);
  {
    int tiles = (NN / 16) * (DD / 16);
    k_wmma_gemm<<<(tiles + 7) / 8, B, 0, stream>>>(hbf, WfT, bf_, h32, NN, DD, FIN);
  }

  const size_t nEdge = (size_t)EE + NN;
  for (int layer = 0; layer < 3; ++layer) {
    k_f2bf<<<gz((size_t)NN * DD), B, 0, stream>>>(h32, hbf, NN * DD);
    int tiles = (NN / 16) * (HD / 16);
    k_wmma_gemm<<<(tiles + 7) / 8, B, 0, stream>>>(hbf, WlT, bl, xl, NN, HD, DD);
    k_wmma_gemm<<<(tiles + 7) / 8, B, 0, stream>>>(hbf, WrT, br, xr, NN, HD, DD);

    k_zero<<<gz((size_t)NN * HH), B, 0, stream>>>((float*)mEnc, (size_t)NN * HH);
    k_zero<<<gz((size_t)NN * HH), B, 0, stream>>>(denom, (size_t)NN * HH);
    k_zero<<<gz((size_t)NN * HD), B, 0, stream>>>(outac, (size_t)NN * HD);

    k_edge_logits<<<gz(nEdge * 32), B, 0, stream>>>(xl, xr, ea, lattr, srcI, dstI, We, att, lw, mEnc);
    k_edge_expw<<<gz(nEdge * HH), B, 0, stream>>>(dstI, lw, mEnc, denom);
    k_edge_scatter<<<gz(nEdge * 32), B, 0, stream>>>(xl, lw, denom, srcI, dstI, outac);

    float* hdst = (layer == 2) ? (float*)d_out : h32;
    k_node<<<gz((size_t)NN * DD), B, 0, stream>>>(outac, bout, hdst);
  }
}
